// DeepseekLayer_13984413516327
// MI455X (gfx1250) — compile-verified, matrix operand order checked by reading, not run on hardware
//
#include <hip/hip_runtime.h>

// ---------------- problem constants ----------------
#define H    2048
#define NH   16
#define HD   128
#define MLPD 8192
#define SEQ  2048
#define EPS  1e-6f
#define QSCALE 0.08838834764831845f   // 1/sqrt(128), folded into RoPE(q)

typedef __attribute__((ext_vector_type(16))) _Float16 v16h;
typedef __attribute__((ext_vector_type(8)))  _Float16 v8h;
typedef __attribute__((ext_vector_type(4)))  _Float16 v4h;
typedef __attribute__((ext_vector_type(8)))  float    v8f;
typedef int v4i __attribute__((vector_size(16)));   // gcc-style, matches builtin param

#define AS1 __attribute__((address_space(1)))
#define AS3 __attribute__((address_space(3)))

#if __has_builtin(__builtin_amdgcn_global_load_async_to_lds_b128)
#define USE_ASYNC 1
#else
#define USE_ASYNC 0
#endif

// 16-byte global -> LDS copy (async DMA path on gfx1250, sync fallback otherwise)
__device__ __forceinline__ void cp16(_Float16* l, const _Float16* g) {
#if USE_ASYNC
    __builtin_amdgcn_global_load_async_to_lds_b128(
        (AS1 v4i*)(unsigned long long)(const void*)g,
        (AS3 v4i*)(unsigned int)(unsigned long long)(const void*)l,
        0, 0);
#else
    *(v8h*)l = *(const v8h*)g;
#endif
}

__device__ __forceinline__ void wait_cp() {
#if USE_ASYNC
#if __has_builtin(__builtin_amdgcn_s_wait_asynccnt)
    __builtin_amdgcn_s_wait_asynccnt(0);
#else
    asm volatile("s_wait_asynccnt 0x0" ::: "memory");
#endif
#endif
}

__device__ __forceinline__ v8f wmma16(v16h a, v16h b, v8f c) {
    return __builtin_amdgcn_wmma_f32_16x16x32_f16(
        false, a, false, b, (short)0, c, false, false);
}

// 16x32 K-major fragment from LDS: two ds_load_b128 per lane.
// Lane L: row = L&15; K-chunks [kb,kb+8) and [kb+16,kb+24), kb=(L>>4)*8.
__device__ __forceinline__ v16h load_frag(const _Float16* p, int stride) {
    const int lane = threadIdx.x & 31;
    const _Float16* base = p + (lane & 15) * stride + ((lane >> 4) << 3);
    const v8h lo = *(const v8h*)(base);
    const v8h hi = *(const v8h*)(base + 16);
    return __builtin_shufflevector(lo, hi, 0,1,2,3,4,5,6,7,8,9,10,11,12,13,14,15);
}

// ---------------- f32 -> f16 bulk convert (weights) ----------------
__global__ __launch_bounds__(256) void cvt_kernel(const float* __restrict__ src,
                                                  _Float16* __restrict__ dst) {
    const size_t i = (size_t)blockIdx.x * 256 + threadIdx.x;   // one float4 each
    const float4 v = ((const float4*)src)[i];
    v4h h;
    h[0] = (_Float16)v.x; h[1] = (_Float16)v.y;
    h[2] = (_Float16)v.z; h[3] = (_Float16)v.w;
    ((v4h*)dst)[i] = h;
}

// ---------------- RMSNorm: f32 in -> f16 out ----------------
__global__ __launch_bounds__(256) void rmsnorm_kernel(const float* __restrict__ x,
                                                      const float* __restrict__ w,
                                                      _Float16* __restrict__ out) {
    __shared__ float red[256];
    const int row = blockIdx.x;
    const float* xr = x + (size_t)row * H;
    float s = 0.f;
#pragma unroll
    for (int i = threadIdx.x; i < H; i += 256) { float v = xr[i]; s += v * v; }
    red[threadIdx.x] = s;
    __syncthreads();
#pragma unroll
    for (int off = 128; off > 0; off >>= 1) {
        if (threadIdx.x < off) red[threadIdx.x] += red[threadIdx.x + off];
        __syncthreads();
    }
    const float rs = rsqrtf(red[0] / (float)H + EPS);
    _Float16* orow = out + (size_t)row * H;
#pragma unroll
    for (int i = threadIdx.x; i < H; i += 256)
        orow[i] = (_Float16)(w[i] * (xr[i] * rs));
}

// ---------------- RoPE in-place on f16 q,k; q pre-scaled by 1/sqrt(HD) -----
__global__ __launch_bounds__(256) void rope_kernel(_Float16* __restrict__ q,
                                                   _Float16* __restrict__ k,
                                                   const float* __restrict__ cosT,
                                                   const float* __restrict__ sinT) {
    int i = blockIdx.x * 256 + threadIdx.x;       // (s, head, d<64)
    int d    = i & 63;
    int rest = i >> 6;
    int head = rest & (NH - 1);
    int s    = rest >> 4;
    const float c1 = cosT[s * HD + d],      s1 = sinT[s * HD + d];
    const float c2 = cosT[s * HD + d + 64], s2 = sinT[s * HD + d + 64];
    const size_t base = (size_t)s * H + head * HD;
    {
        float x1 = (float)q[base + d], x2 = (float)q[base + d + 64];
        q[base + d]      = (_Float16)((x1 * c1 - x2 * s1) * QSCALE);
        q[base + d + 64] = (_Float16)((x2 * c2 + x1 * s2) * QSCALE);
    }
    {
        float x1 = (float)k[base + d], x2 = (float)k[base + d + 64];
        k[base + d]      = (_Float16)(x1 * c1 - x2 * s1);
        k[base + d + 64] = (_Float16)(x2 * c2 + x1 * s2);
    }
}

// ---------------- WMMA GEMM: C[M,N] = epi(A[M,K] . B[N,K]^T) ----------------
// 256x128 block tile, 8 waves as 4(M)x2(N) of 64x64, BK=32, double-buffered
// async LDS staging (DMA for tile k+1 overlaps WMMA on tile k).
// EPI 0: C(f16) = acc       EPI 1: C(f32) = R + acc
// EPI 2: C(f16) = sigmoid(acc) * C   (in-place gate*up)
#define BM 256
#define BN 128
#define BK 32
#define LSTR 40   // padded f16 LDS stride (80 B: 16B-aligned, bank-spread)

__device__ __forceinline__ void stage_panels(_Float16* As, _Float16* Bs,
                                             const _Float16* A, const _Float16* B,
                                             int bm, int bn, int K, int k0, int tid) {
#pragma unroll
    for (int p = 0; p < 4; ++p) {                 // A: 256x32 = 1024 chunks
        const int chunk = tid + p * 256;
        const int r = chunk >> 2;                 // 0..255
        const int c = (chunk & 3) << 3;           // 0,8,16,24
        cp16(&As[r * LSTR + c], &A[(size_t)(bm + r) * K + k0 + c]);
    }
#pragma unroll
    for (int p = 0; p < 2; ++p) {                 // B: 128x32 = 512 chunks
        const int chunk = tid + p * 256;
        const int r = chunk >> 2;                 // 0..127
        const int c = (chunk & 3) << 3;
        cp16(&Bs[r * LSTR + c], &B[(size_t)(bn + r) * K + k0 + c]);
    }
}

template <int EPI, typename CT>
__global__ __launch_bounds__(256) void gemm_atb(const _Float16* __restrict__ A,
                                                const _Float16* __restrict__ B,
                                                CT* __restrict__ C,
                                                const float* __restrict__ R,
                                                int M, int N, int K) {
    __shared__ __attribute__((aligned(16))) _Float16 As[2][BM * LSTR];
    __shared__ __attribute__((aligned(16))) _Float16 Bs[2][BN * LSTR];

    const int tid  = threadIdx.x;
    const int lane = tid & 31;
    const int w    = tid >> 5;       // 0..7
    const int wm   = w >> 1;         // 0..3 : rows wm*64
    const int wn   = w & 1;          // 0..1 : cols wn*64
    const int bm   = blockIdx.y * BM;
    const int bn   = blockIdx.x * BN;

    v8f acc[4][4];
#pragma unroll
    for (int mi = 0; mi < 4; ++mi)
#pragma unroll
        for (int ni = 0; ni < 4; ++ni) acc[mi][ni] = (v8f){0,0,0,0,0,0,0,0};

    stage_panels(As[0], Bs[0], A, B, bm, bn, K, 0, tid);
    wait_cp();
    __syncthreads();

    const int nk = K / BK;
    for (int it = 0; it < nk; ++it) {
        const int cur = it & 1;
        if (it + 1 < nk)   // prefetch next panel into alternate buffer (overlaps WMMA)
            stage_panels(As[cur ^ 1], Bs[cur ^ 1], A, B, bm, bn, K, (it + 1) * BK, tid);

        v16h af[4];
#pragma unroll
        for (int mi = 0; mi < 4; ++mi)
            af[mi] = load_frag(&As[cur][(wm * 64 + mi * 16) * LSTR], LSTR);
#pragma unroll
        for (int ni = 0; ni < 4; ++ni) {
            const v16h bf = load_frag(&Bs[cur][(wn * 64 + ni * 16) * LSTR], LSTR);
#pragma unroll
            for (int mi = 0; mi < 4; ++mi)
                acc[mi][ni] = wmma16(af[mi], bf, acc[mi][ni]);
        }
        wait_cp();
        __syncthreads();
    }

    const int rloc = (lane >> 4) << 3;
    const int cloc = lane & 15;
#pragma unroll
    for (int mi = 0; mi < 4; ++mi)
#pragma unroll
        for (int ni = 0; ni < 4; ++ni)
#pragma unroll
            for (int i = 0; i < 8; ++i) {
                const int row = bm + wm * 64 + mi * 16 + rloc + i;
                const int col = bn + wn * 64 + ni * 16 + cloc;
                const size_t idx = (size_t)row * N + col;
                const float v = acc[mi][ni][i];
                if (EPI == 0)      C[idx] = (CT)v;
                else if (EPI == 1) C[idx] = (CT)(R[idx] + v);
                else               C[idx] = (CT)((1.f / (1.f + __expf(-v))) * (float)C[idx]);
            }
}

// ---------------- Flash attention (full softmax) ----------------
// grid: (SEQ/64, NH); block 256 (8 waves). q/k/v/o are f16; q pre-scaled.
#define QSTR 136   // 272 B
#define VSTR 40
#define SSTR 33
#define PSTR 40

__global__ __launch_bounds__(256) void attn_kernel(const _Float16* __restrict__ q,
                                                   const _Float16* __restrict__ k,
                                                   const _Float16* __restrict__ v,
                                                   _Float16* __restrict__ o) {
    __shared__ __attribute__((aligned(16))) _Float16 Qs[64 * QSTR];
    __shared__ __attribute__((aligned(16))) _Float16 Ks[32 * QSTR];
    __shared__ __attribute__((aligned(16))) _Float16 VTs[HD * VSTR];  // [hd][k]
    __shared__ float    Ssc[64 * SSTR];
    __shared__ __attribute__((aligned(16))) _Float16 Ps[64 * PSTR];
    __shared__ float    mrow[64], lrow[64], arow[64];

    const int tid  = threadIdx.x;
    const int lane = tid & 31;
    const int w    = tid >> 5;
    const int head = blockIdx.y;
    const int q0   = blockIdx.x * 64;
    const size_t hoff = (size_t)head * HD;

    // stage Q: 1024 16B chunks, 4 per thread (async)
#pragma unroll
    for (int p = 0; p < 4; ++p) {
        const int chunk = tid + p * 256;
        const int r = chunk >> 4;           // 0..63
        const int c = (chunk & 15) << 3;    // 0..120
        cp16(&Qs[r * QSTR + c], &q[(size_t)(q0 + r) * H + hoff + c]);
    }
    if (tid < 64) { mrow[tid] = -1e30f; lrow[tid] = 0.f; }

    const int wq = w & 3;   // PV: rows wq*16
    const int wh = w >> 2;  // PV: hd cols wh*64
    const int sq = w & 3;   // scores: tile rows sq*16
    const int sk = w >> 2;  // scores: tile cols sk*16

    v8f oacc[4];
#pragma unroll
    for (int ni = 0; ni < 4; ++ni) oacc[ni] = (v8f){0,0,0,0,0,0,0,0};

    const int rloc = (lane >> 4) << 3;
    const int cloc = lane & 15;

    for (int kb = 0; kb < SEQ; kb += 32) {
        // stage K (async b128 copies) + V transposed (vector load, scalar LDS stores)
#pragma unroll
        for (int p = 0; p < 2; ++p) {
            const int chunk = tid + p * 256;    // 0..511
            const int r = chunk >> 4;           // 0..31
            const int c = (chunk & 15) << 3;    // 0..120
            cp16(&Ks[r * QSTR + c], &k[(size_t)(kb + r) * H + hoff + c]);
            const v8h vv = *(const v8h*)&v[(size_t)(kb + r) * H + hoff + c];
#pragma unroll
            for (int j = 0; j < 8; ++j) VTs[(c + j) * VSTR + r] = vv[j];
        }
        wait_cp();
        __syncthreads();

        // scores: each wave one 16x16 tile of the 64x32 block
        v8f sacc = (v8f){0,0,0,0,0,0,0,0};
#pragma unroll
        for (int kk = 0; kk < 4; ++kk) {
            const v16h aq = load_frag(&Qs[(sq * 16) * QSTR + kk * 32], QSTR);
            const v16h bk = load_frag(&Ks[(sk * 16) * QSTR + kk * 32], QSTR);
            sacc = wmma16(aq, bk, sacc);
        }
#pragma unroll
        for (int i = 0; i < 8; ++i)
            Ssc[(sq * 16 + rloc + i) * SSTR + sk * 16 + cloc] = sacc[i];
        __syncthreads();

        // online softmax per query row
        if (tid < 64) {
            const float mold = mrow[tid];
            float mx = mold;
#pragma unroll
            for (int j = 0; j < 32; ++j) mx = fmaxf(mx, Ssc[tid * SSTR + j]);
            const float al = __expf(mold - mx);
            float sum = 0.f;
#pragma unroll
            for (int j = 0; j < 32; ++j) {
                const float p = __expf(Ssc[tid * SSTR + j] - mx);
                Ps[tid * PSTR + j] = (_Float16)p;
                sum += p;
            }
            lrow[tid] = lrow[tid] * al + sum;
            mrow[tid] = mx;
            arow[tid] = al;
        }
        __syncthreads();

        // rescale accumulators, then P.V
#pragma unroll
        for (int i = 0; i < 8; ++i) {
            const float al = arow[wq * 16 + rloc + i];
#pragma unroll
            for (int ni = 0; ni < 4; ++ni) oacc[ni][i] *= al;
        }
        const v16h ap = load_frag(&Ps[(wq * 16) * PSTR], PSTR);
#pragma unroll
        for (int ni = 0; ni < 4; ++ni) {
            const v16h bv = load_frag(&VTs[(wh * 64 + ni * 16) * VSTR], VSTR);
            oacc[ni] = wmma16(ap, bv, oacc[ni]);
        }
        __syncthreads();
    }

    // normalize and store f16
#pragma unroll
    for (int ni = 0; ni < 4; ++ni)
#pragma unroll
        for (int i = 0; i < 8; ++i) {
            const int row = wq * 16 + rloc + i;
            const int col = wh * 64 + ni * 16 + cloc;
            o[(size_t)(q0 + row) * H + hoff + col] = (_Float16)(oacc[ni][i] / lrow[row]);
        }
}

// ---------------- launch ----------------
extern "C" void kernel_launch(void* const* d_in, const int* in_sizes, int n_in,
                              void* d_out, int out_size, void* d_ws, size_t ws_size,
                              hipStream_t stream) {
    const float* x       = (const float*)d_in[0];
    const float* norm1_w = (const float*)d_in[1];
    const float* wq      = (const float*)d_in[2];
    const float* wk      = (const float*)d_in[3];
    const float* wv      = (const float*)d_in[4];
    const float* wo      = (const float*)d_in[5];
    const float* norm2_w = (const float*)d_in[6];
    const float* w_gate  = (const float*)d_in[7];
    const float* w_up    = (const float*)d_in[8];
    const float* w_down  = (const float*)d_in[9];
    const float* cosT    = (const float*)d_in[10];
    const float* sinT    = (const float*)d_in[11];
    float* out = (float*)d_out;

    // ---- workspace layout (all chunks 16B aligned; hipMalloc base is 256B) ----
    char* p = (char*)d_ws;
    auto fh = [&](size_t n) { _Float16* r = (_Float16*)p; p += n * sizeof(_Float16); return r; };
    auto ff = [&](size_t n) { float* r = (float*)p; p += n * sizeof(float); return r; };

    const size_t HH = (size_t)H * H, MH = (size_t)MLPD * H;
    const size_t SH = (size_t)SEQ * H, SM = (size_t)SEQ * MLPD;
    _Float16* wq16 = fh(HH); _Float16* wk16 = fh(HH);
    _Float16* wv16 = fh(HH); _Float16* wo16 = fh(HH);
    _Float16* wg16 = fh(MH); _Float16* wu16 = fh(MH); _Float16* wd16 = fh(MH);
    _Float16* h16  = fh(SH); _Float16* q16  = fh(SH);
    _Float16* k16  = fh(SH); _Float16* v16  = fh(SH);
    _Float16* t16  = fh(SM);
    float*    x2   = ff(SH);
    _Float16* attn16 = h16;   // h dead after QKV
    _Float16* h2_16  = q16;   // q dead after attention

    const dim3 blk(256);
    const dim3 gH(H / BN, SEQ / BM);
    const dim3 gM(MLPD / BN, SEQ / BM);

    // 0. weights -> f16 (once per launch; O(weights), not O(weights * S/128))
    cvt_kernel<<<HH / 1024, blk, 0, stream>>>(wq,     wq16);
    cvt_kernel<<<HH / 1024, blk, 0, stream>>>(wk,     wk16);
    cvt_kernel<<<HH / 1024, blk, 0, stream>>>(wv,     wv16);
    cvt_kernel<<<HH / 1024, blk, 0, stream>>>(wo,     wo16);
    cvt_kernel<<<MH / 1024, blk, 0, stream>>>(w_gate, wg16);
    cvt_kernel<<<MH / 1024, blk, 0, stream>>>(w_up,   wu16);
    cvt_kernel<<<MH / 1024, blk, 0, stream>>>(w_down, wd16);

    // 1. h = rmsnorm(x)
    rmsnorm_kernel<<<SEQ, blk, 0, stream>>>(x, norm1_w, h16);
    // 2-4. q,k,v
    gemm_atb<0, _Float16><<<gH, blk, 0, stream>>>(h16, wq16, q16, nullptr, SEQ, H, H);
    gemm_atb<0, _Float16><<<gH, blk, 0, stream>>>(h16, wk16, k16, nullptr, SEQ, H, H);
    gemm_atb<0, _Float16><<<gH, blk, 0, stream>>>(h16, wv16, v16, nullptr, SEQ, H, H);
    // 5. RoPE (q scaled by 1/sqrt(HD))
    rope_kernel<<<(SEQ * NH * 64) / 256, blk, 0, stream>>>(q16, k16, cosT, sinT);
    // 6. attention
    attn_kernel<<<dim3(SEQ / 64, NH), blk, 0, stream>>>(q16, k16, v16, attn16);
    // 7. x2 = x + attn . Wo^T
    gemm_atb<1, float><<<gH, blk, 0, stream>>>(attn16, wo16, x2, x, SEQ, H, H);
    // 8. h2 = rmsnorm(x2)
    rmsnorm_kernel<<<SEQ, blk, 0, stream>>>(x2, norm2_w, h2_16);
    // 9. t = h2 . Wup^T ; 10. t = sigmoid(h2 . Wgate^T) * t   (in place)
    gemm_atb<0, _Float16><<<gM, blk, 0, stream>>>(h2_16, wu16, t16, nullptr, SEQ, MLPD, H);
    gemm_atb<2, _Float16><<<gM, blk, 0, stream>>>(h2_16, wg16, t16, nullptr, SEQ, MLPD, H);
    // 11. out = x2 + t . Wdown^T
    gemm_atb<1, float><<<gH, blk, 0, stream>>>(t16, wd16, out, x2, SEQ, H, MLPD);
}
